// CausalSelfAttention_32684701122571
// MI455X (gfx1250) — compile-verified, matrix operand order checked by reading, not run on hardware
//
#include <hip/hip_runtime.h>
#include <hip/hip_bf16.h>

// CausalSelfAttention for MI455X (gfx1250, wave32, WMMA).
// B=4, T=2048, D=1024, H=16, HD=64.

#if defined(__has_builtin)
#if __has_builtin(__builtin_amdgcn_global_load_async_to_lds_b128) && \
    __has_builtin(__builtin_amdgcn_s_wait_asynccnt)
#define USE_ASYNC_LDS 1
#endif
#endif
#ifndef USE_ASYNC_LDS
#define USE_ASYNC_LDS 0
#endif

namespace {

constexpr int Bc  = 4;
constexpr int Tc  = 2048;
constexpr int Dc  = 1024;
constexpr int Hc  = 16;
constexpr int HDc = 64;
constexpr int D3c = 3072;

// Padded LDS row stride for B tiles: 40 bf16 = 80 B = 20 dwords.
// 16-row fragment reads then hit 16 distinct banks (20*r mod 64 all distinct).
constexpr int BSTR = 40;

typedef __bf16 bf16_t;
typedef __attribute__((ext_vector_type(16))) bf16_t v16bf;
typedef __attribute__((ext_vector_type(8)))  bf16_t v8bf;
typedef __attribute__((ext_vector_type(8)))  float  v8f;
typedef __attribute__((ext_vector_type(4)))  float  v4f;

__device__ inline v8f wmma_bf16(v16bf a, v16bf b, v8f c) {
  // D(f32 16x16) = A(bf16 16x32) * B(bf16 32x16) + C
  return __builtin_amdgcn_wmma_f32_16x16x32_bf16(
      /*neg_a=*/false, a, /*neg_b=*/false, b,
      /*c_mod=*/(short)0, c, /*reuse_a=*/false, /*reuse_b=*/false);
}

// A/B fragment from bf16 row data (global or LDS; clang infers the address
// space after inlining -> global_load_b128 or ds_load_b128).
// Element mapping (16-bit A 16x32, ISA 7.12.2):
//   lanes 0-15 : e0..7 -> K=0..7,   e8..15 -> K=16..23
//   lanes 16-31: e0..7 -> K=8..15,  e8..15 -> K=24..31
// i.e. per lane two contiguous 16B runs at +0 and +16 elems (after +half*8).
__device__ inline v16bf frag_bf16(const bf16_t* p) {
  v8bf lo = *(const v8bf*)(p);
  v8bf hi = *(const v8bf*)(p + 16);
  return __builtin_shufflevector(lo, hi,
      0, 1, 2, 3, 4, 5, 6, 7, 8, 9, 10, 11, 12, 13, 14, 15);
}

// ---------------------------------------------------------------------------
// Kernel 0: one-shot fp32 -> bf16 conversion (x, w_qkv, w_out).
// ---------------------------------------------------------------------------
__global__ __launch_bounds__(256) void cvt_kernel(const float* __restrict__ in,
                                                  bf16_t* __restrict__ out,
                                                  int n) {
  const int i = (blockIdx.x * 256 + threadIdx.x) * 8;
  if (i >= n) return;
  v4f a = *(const v4f*)(in + i);
  v4f b = *(const v4f*)(in + i + 4);
  v8bf o;
#pragma unroll
  for (int j = 0; j < 4; ++j) {
    o[j]     = (bf16_t)a[j];
    o[j + 4] = (bf16_t)b[j];
  }
  *(v8bf*)(out + i) = o;
}

// ---------------------------------------------------------------------------
// Block-tiled GEMM core: block owns a 128x64 output tile (8 waves x 16 rows,
// all waves share the 64-column B strip). B k-slices (64x32 bf16, 4 KB) are
// staged in double-buffered LDS: each of the 256 threads moves exactly 16 B
// per k-step via the CDNA5 async global->LDS path (ASYNCcnt) where available.
// A fragments stay register-direct (rows are private per wave).
// Traffic: 48 B/output element vs 160 B/element for the unshared version.
// ---------------------------------------------------------------------------
#if USE_ASYNC_LDS
typedef int v4i_t __attribute__((ext_vector_type(4)));
typedef __attribute__((address_space(1))) v4i_t* gaddr4_t;  // global 16B packet
typedef __attribute__((address_space(3))) v4i_t* laddr4_t;  // LDS 16B packet
#endif

template <typename EPI>
__device__ inline void gemm_block_64(const bf16_t* __restrict__ pa,
                                     const bf16_t* __restrict__ pbg,
                                     bf16_t (*Bs)[64][BSTR],
                                     int brow, int bseg, int r, int hf,
                                     EPI&& epilogue) {
  // Stage k-slice 0.
  *(v8bf*)&Bs[0][brow][bseg] = *(const v8bf*)(pbg);
  __syncthreads();

  v8f acc0 = {}, acc1 = {}, acc2 = {}, acc3 = {};
  int cur = 0;
  for (int kb = 0; kb < Dc; kb += 32) {
    const int  nxt  = cur ^ 1;
    const bool more = (kb + 32) < Dc;
#if USE_ASYNC_LDS
    if (more)
      __builtin_amdgcn_global_load_async_to_lds_b128(
          (gaddr4_t)(pbg + kb + 32), (laddr4_t)&Bs[nxt][brow][bseg], 0, 0);
#else
    v8bf tmp = {};
    if (more) tmp = *(const v8bf*)(pbg + kb + 32);  // overlaps the WMMAs below
#endif
    const v16bf a = frag_bf16(pa + kb);
    acc0 = wmma_bf16(a, frag_bf16(&Bs[cur][0 * 16 + r][hf * 8]), acc0);
    acc1 = wmma_bf16(a, frag_bf16(&Bs[cur][1 * 16 + r][hf * 8]), acc1);
    acc2 = wmma_bf16(a, frag_bf16(&Bs[cur][2 * 16 + r][hf * 8]), acc2);
    acc3 = wmma_bf16(a, frag_bf16(&Bs[cur][3 * 16 + r][hf * 8]), acc3);
#if USE_ASYNC_LDS
    if (more) __builtin_amdgcn_s_wait_asynccnt(0);
#else
    if (more) *(v8bf*)&Bs[nxt][brow][bseg] = tmp;
#endif
    __syncthreads();
    cur = nxt;
  }
  epilogue(acc0, acc1, acc2, acc3);
}

// ---------------------------------------------------------------------------
// Kernel 1: qkv = x @ w_qkv^T  (M=8192, N=3072, K=1024), bf16 inputs.
// Epilogue scatters into attention layouts:
//   Q,K: [B,H,T,HD] bf16 (Q pre-scaled by 1/sqrt(HD))
//   Vt : [B,H,HD,T] bf16 (transposed so PV B-fragments are contiguous loads)
// ---------------------------------------------------------------------------
__global__ __launch_bounds__(256) void qkv_kernel(const bf16_t* __restrict__ xb,
                                                  const bf16_t* __restrict__ wqb,
                                                  bf16_t* __restrict__ Q,
                                                  bf16_t* __restrict__ K,
                                                  bf16_t* __restrict__ Vt) {
  __shared__ __align__(16) bf16_t Bs[2][64][BSTR];  // double-buffered B tile

  const int tid  = threadIdx.x;
  const int lane = tid & 31;
  const int r    = lane & 15;
  const int hf   = lane >> 4;
  const int wv   = tid >> 5;
  const int NGB  = D3c / 64;                 // 48 column blocks
  const int mt0  = (blockIdx.x / NGB) * 128; // block row base
  const int n0   = (blockIdx.x % NGB) * 64;  // block col base

  const bf16_t* pa = xb + (size_t)(mt0 + wv * 16 + r) * Dc + hf * 8;
  // Cooperative B staging: thread -> (row, 16B segment).
  const int brow = tid >> 2;
  const int bseg = (tid & 3) * 8;
  const bf16_t* pbg = wqb + (size_t)(n0 + brow) * Dc + bseg;

  gemm_block_64(pa, pbg, Bs, brow, bseg, r, hf,
      [&](v8f acc0, v8f acc1, v8f acc2, v8f acc3) {
    v8f accs[4] = {acc0, acc1, acc2, acc3};
#pragma unroll
    for (int j = 0; j < 4; ++j) {
      const int ncol  = n0 + j * 16 + r;
      const int which = ncol / Dc;   // 0=q 1=k 2=v
      const int dloc  = ncol % Dc;
      const int hh    = dloc / HDc;
      const int hd    = dloc % HDc;
#pragma unroll
      for (int v = 0; v < 8; ++v) {
        const int m  = mt0 + wv * 16 + v + 8 * hf;
        const int bb = m / Tc, t = m % Tc;
        const float val = accs[j][v];
        if (which == 0) {
          Q[((size_t)(bb * Hc + hh) * Tc + t) * HDc + hd] = (bf16_t)(val * 0.125f);
        } else if (which == 1) {
          K[((size_t)(bb * Hc + hh) * Tc + t) * HDc + hd] = (bf16_t)val;
        } else {
          Vt[((size_t)(bb * Hc + hh) * HDc + hd) * Tc + t] = (bf16_t)val;
        }
      }
    }
  });
}

// ---------------------------------------------------------------------------
// Kernel 2: flash attention. One wave per (b, h, 16-row q tile).
// Online softmax over 32-key chunks; K frags ahead of S-WMMAs, V frags issued
// before the softmax VALU section, LDS bounce re-layouts P (D-frag -> A-frag).
// ---------------------------------------------------------------------------
__global__ __launch_bounds__(256) void attn_kernel(const bf16_t* __restrict__ Q,
                                                   const bf16_t* __restrict__ K,
                                                   const bf16_t* __restrict__ Vt,
                                                   bf16_t* __restrict__ AO) {
  __shared__ __align__(16) bf16_t pls[8][16][32];  // per-wave P tile (8 KB)

  const int lane = threadIdx.x & 31;
  const int r    = lane & 15;
  const int hf   = lane >> 4;
  const int wv   = threadIdx.x >> 5;
  const int wid  = blockIdx.x * 8 + wv;
  const int qt   = wid & 127;      // 128 q tiles per (b,h)
  const int bh   = wid >> 7;       // b*H + h
  const int bb   = bh >> 4;
  const int hh   = bh & 15;

  const bf16_t* Qb = Q  + (size_t)bh * Tc * HDc;
  const bf16_t* Kb = K  + (size_t)bh * Tc * HDc;
  const bf16_t* Vb = Vt + (size_t)bh * HDc * Tc;

  const v16bf qf0 = frag_bf16(Qb + (size_t)(qt * 16 + r) * HDc + 0  + hf * 8);
  const v16bf qf1 = frag_bf16(Qb + (size_t)(qt * 16 + r) * HDc + 32 + hf * 8);

  float mrow[8], lrow[8];
  v8f o0 = {}, o1 = {}, o2 = {}, o3 = {};
#pragma unroll
  for (int v = 0; v < 8; ++v) { mrow[v] = -__builtin_inff(); lrow[v] = 0.0f; }

  const int qend = qt * 16 + 15;
  for (int k0 = 0; k0 <= qend; k0 += 32) {
    const v16bf kf00 = frag_bf16(Kb + (size_t)(k0 + r)      * HDc + 0  + hf * 8);
    const v16bf kf01 = frag_bf16(Kb + (size_t)(k0 + r)      * HDc + 32 + hf * 8);
    const v16bf kf10 = frag_bf16(Kb + (size_t)(k0 + 16 + r) * HDc + 0  + hf * 8);
    const v16bf kf11 = frag_bf16(Kb + (size_t)(k0 + 16 + r) * HDc + 32 + hf * 8);
    // V loads issued early so they overlap the softmax VALU section.
    const v16bf vf0 = frag_bf16(Vb + (size_t)(0 * 16 + r) * Tc + k0 + hf * 8);
    const v16bf vf1 = frag_bf16(Vb + (size_t)(1 * 16 + r) * Tc + k0 + hf * 8);
    const v16bf vf2 = frag_bf16(Vb + (size_t)(2 * 16 + r) * Tc + k0 + hf * 8);
    const v16bf vf3 = frag_bf16(Vb + (size_t)(3 * 16 + r) * Tc + k0 + hf * 8);

    v8f s0 = {}, s1 = {};
    s0 = wmma_bf16(qf0, kf00, s0);
    s0 = wmma_bf16(qf1, kf01, s0);
    s1 = wmma_bf16(qf0, kf10, s1);
    s1 = wmma_bf16(qf1, kf11, s1);

    const int kc0 = k0 + r;
    const int kc1 = k0 + 16 + r;
#pragma unroll
    for (int v = 0; v < 8; ++v) {
      const int qrow = qt * 16 + v + 8 * hf;
      float a = (kc0 > qrow) ? -__builtin_inff() : s0[v];
      float b = (kc1 > qrow) ? -__builtin_inff() : s1[v];
      float cm = fmaxf(a, b);
#pragma unroll
      for (int mm = 8; mm >= 1; mm >>= 1) cm = fmaxf(cm, __shfl_xor(cm, mm, 32));
      const float mn    = fmaxf(mrow[v], cm);
      const float alpha = __expf(mrow[v] - mn);
      const float p0    = __expf(a - mn);
      const float p1    = __expf(b - mn);
      float rs = p0 + p1;
#pragma unroll
      for (int mm = 8; mm >= 1; mm >>= 1) rs += __shfl_xor(rs, mm, 32);
      lrow[v] = lrow[v] * alpha + rs;
      mrow[v] = mn;
      o0[v] *= alpha; o1[v] *= alpha; o2[v] *= alpha; o3[v] *= alpha;
      pls[wv][v + 8 * hf][r]      = (bf16_t)p0;
      pls[wv][v + 8 * hf][16 + r] = (bf16_t)p1;
    }
    __asm__ volatile("s_wait_dscnt 0x0" ::: "memory");

    const v16bf pf = frag_bf16(&pls[wv][r][hf * 8]);
    o0 = wmma_bf16(pf, vf0, o0);
    o1 = wmma_bf16(pf, vf1, o1);
    o2 = wmma_bf16(pf, vf2, o2);
    o3 = wmma_bf16(pf, vf3, o3);
    __asm__ volatile("s_wait_dscnt 0x0" ::: "memory");
  }

#pragma unroll
  for (int v = 0; v < 8; ++v) {
    const float inv = 1.0f / lrow[v];
    const int t = qt * 16 + v + 8 * hf;
    bf16_t* dst = AO + (size_t)(bb * Tc + t) * Dc + hh * HDc;
    dst[0 * 16 + r] = (bf16_t)(o0[v] * inv);
    dst[1 * 16 + r] = (bf16_t)(o1[v] * inv);
    dst[2 * 16 + r] = (bf16_t)(o2[v] * inv);
    dst[3 * 16 + r] = (bf16_t)(o3[v] * inv);
  }
}

// ---------------------------------------------------------------------------
// Kernel 3: out = AO @ w_out^T  (M=8192, N=1024, K=1024), bf16 in, fp32 out.
// Same block-tiled structure as kernel 1.
// ---------------------------------------------------------------------------
__global__ __launch_bounds__(256) void proj_kernel(const bf16_t* __restrict__ AO,
                                                   const bf16_t* __restrict__ wob,
                                                   float* __restrict__ out) {
  __shared__ __align__(16) bf16_t Bs[2][64][BSTR];

  const int tid  = threadIdx.x;
  const int lane = tid & 31;
  const int r    = lane & 15;
  const int hf   = lane >> 4;
  const int wv   = tid >> 5;
  const int NGB  = Dc / 64;                  // 16 column blocks
  const int mt0  = (blockIdx.x / NGB) * 128;
  const int n0   = (blockIdx.x % NGB) * 64;

  const bf16_t* pa = AO + (size_t)(mt0 + wv * 16 + r) * Dc + hf * 8;
  const int brow = tid >> 2;
  const int bseg = (tid & 3) * 8;
  const bf16_t* pbg = wob + (size_t)(n0 + brow) * Dc + bseg;

  gemm_block_64(pa, pbg, Bs, brow, bseg, r, hf,
      [&](v8f acc0, v8f acc1, v8f acc2, v8f acc3) {
    v8f accs[4] = {acc0, acc1, acc2, acc3};
#pragma unroll
    for (int j = 0; j < 4; ++j) {
#pragma unroll
      for (int v = 0; v < 8; ++v) {
        const int m = mt0 + wv * 16 + v + 8 * hf;
        out[(size_t)m * Dc + n0 + j * 16 + r] = accs[j][v];
      }
    }
  });
}

}  // namespace

extern "C" void kernel_launch(void* const* d_in, const int* in_sizes, int n_in,
                              void* d_out, int out_size, void* d_ws, size_t ws_size,
                              hipStream_t stream) {
  const float* x     = (const float*)d_in[0];
  const float* w_qkv = (const float*)d_in[1];
  const float* w_out = (const float*)d_in[2];
  float* out = (float*)d_out;

  // Workspace layout (bf16 elements):
  //   Q, K, Vt, AO : 4 x B*H*T*HD = 4 x 8388608   (64 MB)
  //   xb           : B*T*D       = 8388608        (16 MB)
  //   wqb          : 3D*D        = 3145728        ( 6 MB)
  //   wob          : D*D         = 1048576        ( 2 MB)
  const size_t perMat = (size_t)Bc * Hc * Tc * HDc;  // 8388608
  bf16_t* Q   = (bf16_t*)d_ws;
  bf16_t* K   = Q + perMat;
  bf16_t* Vt  = K + perMat;
  bf16_t* AO  = Vt + perMat;
  bf16_t* xb  = AO + perMat;
  bf16_t* wqb = xb + perMat;
  bf16_t* wob = wqb + (size_t)D3c * Dc;

  const int nx = Bc * Tc * Dc;   // 8388608
  const int nq = D3c * Dc;       // 3145728
  const int no = Dc * Dc;        // 1048576

  cvt_kernel<<<nx / (8 * 256), 256, 0, stream>>>(x, xb, nx);
  cvt_kernel<<<nq / (8 * 256), 256, 0, stream>>>(w_qkv, wqb, nq);
  cvt_kernel<<<no / (8 * 256), 256, 0, stream>>>(w_out, wob, no);

  // Block = 128x64 output tile (8 waves), B strip shared through LDS.
  qkv_kernel<<<(8192 / 128) * (D3c / 64), 256, 0, stream>>>(xb, wqb, Q, K, Vt);
  attn_kernel<<<(Bc * Hc * 128) / 8, 256, 0, stream>>>(Q, K, Vt, AO);
  proj_kernel<<<(8192 / 128) * (Dc / 64), 256, 0, stream>>>(AO, wob, out);
}
